// U_R_Aggregation_12283606466575
// MI455X (gfx1250) — matrix operand order repaired and checked by reading, not compiled, not run on hardware
//
#include <hip/hip_runtime.h>
#include <hip/hip_bf16.h>
#include <math.h>

// ---------------------------------------------------------------------------
// Fused U_R_Aggregation for MI455X (gfx1250, wave32, WMMA f16->f32)
//   B=4096, L=50 (padded to 64 rows), D=64
//   Prep kernel: convert all weights f32->f16 once into d_ws.
//   Main kernel: 2 batch elements per block, 256 threads = 8 waves.
//     waves 0-3 -> b0 rows (16 rows each), waves 4-7 -> b1 rows.
//     One shared 16KB LDS weight image per layer (f16 copy from L2).
// ---------------------------------------------------------------------------

typedef __attribute__((ext_vector_type(16))) _Float16 v16h;
typedef __attribute__((ext_vector_type(8)))  _Float16 v8h;
typedef __attribute__((ext_vector_type(8)))  float    v8f;

#define LHIST 50
#define LP    64      // padded history rows per batch element
#define BPB   2       // batch elements per block
#define XS    136     // f16 row stride (128 + 8 pad)

// f16 weight image offsets inside d_ws (in halfs)
#define WS_WR1   0        // 64*128
#define WS_WR2   8192     // 64*64
#define WS_ATT1  12288    // 64*128
#define WS_ATT2  20480    // 64*64
#define WS_ATT3  24576    // 64
#define WS_TOTAL 24640

__device__ __forceinline__ v16h combine16(v8h lo, v8h hi) {
  return __builtin_shufflevector(lo, hi, 0,1,2,3,4,5,6,7,8,9,10,11,12,13,14,15);
}

// A fragment (16x32 f16): lane half=lane>>4 supplies k = kbase+half*16+{0..15} in
// elements 0..15; row = lane&15.
__device__ __forceinline__ v16h load_a_frag(const _Float16* base, int kbase, int lane) {
  const int half = lane >> 4, row = lane & 15;
  const _Float16* p = base + row * XS + kbase + half * 16;
  v8h lo = *(const v8h*)(p);
  v8h hi = *(const v8h*)(p + 8);
  return combine16(lo, hi);
}

// B fragment (32x16 f16): lane n = lane&15 (+ntile*16), element j supplies
// k = kbase + (lane>>4)*16 + j. Weights stored row-major [n][K].
__device__ __forceinline__ v16h load_b_frag(const _Float16* w, int K, int kbase, int ntile, int lane) {
  const int khalf = lane >> 4;
  const int n = (lane & 15) + ntile * 16;
  return *(const v16h*)(w + n * K + kbase + khalf * 16);
}

// One wave: 16x64 tile, out = relu(act[16xK] @ W^T + bias) stored f16 to LDS.
template <int K>
__device__ __forceinline__ void wave_gemm_relu(const _Float16* act, const _Float16* w,
                                               const float* bias, _Float16* dst, int lane) {
  v8f acc[4];
  const v8f zero = {0.f, 0.f, 0.f, 0.f, 0.f, 0.f, 0.f, 0.f};
#pragma unroll
  for (int nt = 0; nt < 4; ++nt) acc[nt] = zero;

#pragma unroll
  for (int ks = 0; ks < K; ks += 32) {
    v16h a = load_a_frag(act, ks, lane);
#pragma unroll
    for (int nt = 0; nt < 4; ++nt) {
      v16h b = load_b_frag(w, K, ks, nt, lane);
      acc[nt] = __builtin_amdgcn_wmma_f32_16x16x32_f16(
          false, a, false, b, (short)0, acc[nt], false, false);
    }
  }

  const int n0 = lane & 15, half = lane >> 4;
#pragma unroll
  for (int nt = 0; nt < 4; ++nt) {
    float bv = bias[nt * 16 + n0];
#pragma unroll
    for (int r = 0; r < 8; ++r) {
      float v = acc[nt][r] + bv;
      v = v > 0.f ? v : 0.f;
      dst[(r + 8 * half) * XS + nt * 16 + n0] = (_Float16)v;
    }
  }
}

// ---- Prep: convert weights to f16 once (tiny; runs on every launch) -------
__global__ void weight_conv_kernel(const float* __restrict__ w_r1, const float* __restrict__ w_r2,
                                   const float* __restrict__ att1_w, const float* __restrict__ att2_w,
                                   const float* __restrict__ att3_w, _Float16* __restrict__ dst) {
  int i = blockIdx.x * blockDim.x + threadIdx.x;
  if (i < WS_WR2)        dst[i] = (_Float16)w_r1[i - WS_WR1];
  else if (i < WS_ATT1)  dst[i] = (_Float16)w_r2[i - WS_WR2];
  else if (i < WS_ATT2)  dst[i] = (_Float16)att1_w[i - WS_ATT1];
  else if (i < WS_ATT3)  dst[i] = (_Float16)att2_w[i - WS_ATT2];
  else if (i < WS_TOTAL) dst[i] = (_Float16)att3_w[i - WS_ATT3];
}

__device__ __forceinline__ void copy_w(void* dst, const void* src, int bytes, int tid) {
  uint4* d = (uint4*)dst;
  const uint4* s = (const uint4*)src;
  for (int i = tid; i < bytes / 16; i += 256) d[i] = s[i];
}

__global__ __launch_bounds__(256, 1)
void urag_kernel(const int* __restrict__ nodes, const int* __restrict__ history,
                 const int* __restrict__ ratings, const int* __restrict__ lengths,
                 const float* __restrict__ u2e, const float* __restrict__ r2e,
                 const float* __restrict__ rating2e,
                 const _Float16* __restrict__ wsh,     // f16 weight image
                 const float* __restrict__ b_r1, const float* __restrict__ b_r2,
                 const float* __restrict__ att1_b, const float* __restrict__ att2_b,
                 const float* __restrict__ att3_b,
                 float* __restrict__ out, int nbatch) {
  __shared__ __align__(16) _Float16 xbuf[BPB * LP * XS];   // x_in / (o|urep) / (o|a2)
  __shared__ __align__(16) _Float16 hbuf[BPB * LP * XS];   // h1 / a1
  __shared__ __align__(16) _Float16 wlds[64 * 128];        // current layer's weights
  __shared__ float    biaslds[64];
  __shared__ _Float16 ubuf[BPB][64];
  __shared__ _Float16 a3lds[64];
  __shared__ float    scorebuf[BPB * LP];
  __shared__ float    wght[BPB * LP];

  const int b0   = blockIdx.x * BPB;
  const int tid  = threadIdx.x;
  const int lane = tid & 31;
  const int wave = tid >> 5;        // 0..7
  const int sub  = wave >> 2;       // which batch element (0/1)
  const int rowb = (sub * LP) + (wave & 3) * 16;

  // ---- Stage A: weights L1 + gathers -------------------------------------
  copy_w(wlds, wsh + WS_WR1, 64 * 128 * 2, tid);
  if (tid < 64) { biaslds[tid] = b_r1[tid]; a3lds[tid] = wsh[WS_ATT3 + tid]; }
  if (tid >= 64 && tid < 64 + BPB * 64) {
    int t = tid - 64, s = t >> 6, c = t & 63, bb = b0 + s;
    ubuf[s][c] = (bb < nbatch) ? (_Float16)u2e[(long long)nodes[bb] * 64 + c] : (_Float16)0.f;
  }
  for (int i = tid; i < BPB * LP * 64; i += 256) {
    int row = i >> 6, col = i & 63;          // row in 0..127
    int s = row >> 6, l = row & 63, bb = b0 + s;
    float vu = 0.f, vr = 0.f;
    if (l < LHIST && bb < nbatch) {
      int hidx = history[bb * LHIST + l];
      int ridx = ratings[bb * LHIST + l];
      vu = r2e[(long long)hidx * 64 + col];
      vr = rating2e[(long long)ridx * 64 + col];
    }
    xbuf[row * XS + col]      = (_Float16)vu;   // e_ur
    xbuf[row * XS + 64 + col] = (_Float16)vr;   // e_rating
  }
  __syncthreads();

  // ---- GEMM1: h1 = relu(x @ w_r1^T + b_r1) -------------------------------
  wave_gemm_relu<128>(xbuf + rowb * XS, wlds, biaslds, hbuf + rowb * XS, lane);
  __syncthreads();

  // ---- Stage B: weights L2, broadcast urep into xbuf upper half ----------
  copy_w(wlds, wsh + WS_WR2, 64 * 64 * 2, tid);
  if (tid < 64) biaslds[tid] = b_r2[tid];
  for (int i = tid; i < BPB * LP * 64; i += 256) {
    int row = i >> 6, col = i & 63, s = row >> 6;
    xbuf[row * XS + 64 + col] = ubuf[s][col];
  }
  __syncthreads();

  // ---- GEMM2: o = relu(h1 @ w_r2^T + b_r2) -> xbuf[:, 0:64] --------------
  wave_gemm_relu<64>(hbuf + rowb * XS, wlds, biaslds, xbuf + rowb * XS, lane);
  __syncthreads();

  // ---- Stage C: att1 weights ---------------------------------------------
  copy_w(wlds, wsh + WS_ATT1, 64 * 128 * 2, tid);
  if (tid < 64) biaslds[tid] = att1_b[tid];
  __syncthreads();

  // ---- GEMM3: a1 = relu([o|urep] @ att1^T + b) -> hbuf -------------------
  wave_gemm_relu<128>(xbuf + rowb * XS, wlds, biaslds, hbuf + rowb * XS, lane);
  __syncthreads();

  // ---- Stage D: att2 weights ---------------------------------------------
  copy_w(wlds, wsh + WS_ATT2, 64 * 64 * 2, tid);
  if (tid < 64) biaslds[tid] = att2_b[tid];
  __syncthreads();

  // ---- GEMM4: a2 = relu(a1 @ att2^T + b) -> xbuf[:, 64:128] --------------
  wave_gemm_relu<64>(hbuf + rowb * XS, wlds, biaslds, xbuf + rowb * XS + 64, lane);
  __syncthreads();

  // ---- Scores: s_l = a2 . att3 + b3, ragged mask -------------------------
  if (tid < BPB * LP) {
    int s = tid >> 6, l = tid & 63, bb = b0 + s;
    int len = (bb < nbatch) ? lengths[bb] : 0;
    float sc = att3_b[0];
    const _Float16* a2 = xbuf + tid * XS + 64;
#pragma unroll
    for (int c = 0; c < 64; ++c) sc += (float)a2[c] * (float)a3lds[c];
    scorebuf[tid] = (l < len) ? sc : -1e30f;   // len <= 50 <= LP, pads masked
  }
  __syncthreads();

  // ---- Stable softmax per batch element (cheap: 64 elems each) -----------
  if (tid < BPB) {
    const float* sb = scorebuf + tid * LP;
    float* wb = wght + tid * LP;
    float m = -1e30f;
    for (int l = 0; l < LP; ++l) m = fmaxf(m, sb[l]);
    float ssum = 0.f;
    for (int l = 0; l < LP; ++l) { float e = __expf(sb[l] - m); wb[l] = e; ssum += e; }
    float inv = 1.f / ssum;
    for (int l = 0; l < LP; ++l) wb[l] *= inv;
  }
  __syncthreads();

  // ---- out[b, d] = sum_l w_l * o[l, d] (o is f16 in xbuf[:, 0:64]) -------
  if (tid < BPB * 64) {
    int s = tid >> 6, d = tid & 63, bb = b0 + s;
    if (bb < nbatch) {
      const _Float16* ob = xbuf + s * LP * XS;
      const float* wb = wght + s * LP;
      float acc = 0.f;
      for (int l = 0; l < LP; ++l) acc += wb[l] * (float)ob[l * XS + d];
      out[(size_t)bb * 64 + d] = acc;
    }
  }
}

extern "C" void kernel_launch(void* const* d_in, const int* in_sizes, int n_in,
                              void* d_out, int out_size, void* d_ws, size_t ws_size,
                              hipStream_t stream) {
  const int*   nodes    = (const int*)d_in[0];
  const int*   history  = (const int*)d_in[1];
  const int*   ratings  = (const int*)d_in[2];
  const int*   lengths  = (const int*)d_in[3];
  const float* u2e      = (const float*)d_in[4];
  const float* r2e      = (const float*)d_in[5];
  const float* rating2e = (const float*)d_in[6];
  const float* w_r1     = (const float*)d_in[7];
  const float* b_r1     = (const float*)d_in[8];
  const float* w_r2     = (const float*)d_in[9];
  const float* b_r2     = (const float*)d_in[10];
  const float* att1_w   = (const float*)d_in[11];
  const float* att1_b   = (const float*)d_in[12];
  const float* att2_w   = (const float*)d_in[13];
  const float* att2_b   = (const float*)d_in[14];
  const float* att3_w   = (const float*)d_in[15];
  const float* att3_b   = (const float*)d_in[16];
  float* out = (float*)d_out;
  _Float16* wsh = (_Float16*)d_ws;

  // One-time (per launch) f32->f16 weight image build in workspace.
  weight_conv_kernel<<<(WS_TOTAL + 255) / 256, 256, 0, stream>>>(
      w_r1, w_r2, att1_w, att2_w, att3_w, wsh);

  const int B = in_sizes[0];  // 4096
  const int nblocks = (B + BPB - 1) / BPB;
  urag_kernel<<<nblocks, 256, 0, stream>>>(nodes, history, ratings, lengths,
                                           u2e, r2e, rating2e, wsh,
                                           b_r1, b_r2, att1_b, att2_b, att3_b,
                                           out, B);
}
